// D3MCELL_57990648430623
// MI455X (gfx1250) — compile-verified, matrix operand order checked by reading, not run on hardware
//
#include <hip/hip_runtime.h>

// ---------------------------------------------------------------------------
// Types for WMMA (gfx1250, wave32): D(16x16 f32) = A(16x32 bf16) x B(32x16 bf16) + C
// ---------------------------------------------------------------------------
typedef unsigned int u32x4 __attribute__((ext_vector_type(4)));
typedef __bf16 v16bf __attribute__((ext_vector_type(16)));
typedef float  v8f   __attribute__((ext_vector_type(8)));

union FragAB { v16bf v; u32x4 q[2]; };
union FragC  { v8f v; float f[8]; };

__device__ __forceinline__ __bf16 f2bf(float f) {
  union { float f; unsigned u; } in; in.f = f;
  unsigned r = in.u + 0x7FFFu + ((in.u >> 16) & 1u);   // round-to-nearest-even
  union { unsigned short s; __bf16 b; } out; out.s = (unsigned short)(r >> 16);
  return out.b;
}

__device__ __forceinline__ float sigf(float x) { return 1.0f / (1.0f + __expf(-x)); }

// --- CDNA5 async global->LDS copy (ASYNCcnt-tracked), via inline asm --------
// LDS destination address = low 32 bits of the generic pointer (ISA 10.2).
// INST_OFFSET is added to BOTH the LDS and global addresses (ISA 8 4.4), so a
// 32B chunk is two instructions sharing one address register via offset:16.
__device__ __forceinline__ void async_ld_b128_pair(unsigned lds_addr, const void* gaddr) {
  asm volatile("global_load_async_to_lds_b128 %0, %1, off\n\t"
               "global_load_async_to_lds_b128 %0, %1, off offset:16"
               :: "v"(lds_addr), "v"(gaddr) : "memory");
}
__device__ __forceinline__ void s_wait_async0() {
  asm volatile("s_wait_asynccnt 0x0" ::: "memory");
}
__device__ __forceinline__ void s_wait_async4() {
  asm volatile("s_wait_asynccnt 0x4" ::: "memory");
}

// ---------------------------------------------------------------------------
// Generic batched bf16 GEMM:  C[bz] (M x N, f32) (+)= act[bz] (M x K) * wgt[bz] (N x K)^T + bias
//   - act/wgt bf16, row-major, K contiguous. M,N multiples of 128; K multiple of 32.
//   - beta=1: accumulate into existing C.  bias may be null (per-output-column f32).
// Workgroup: 256 threads = 8 wave32, 128x128 C tile, each wave 32(M)x64(N) = 8 WMMA accums.
// Staging: double-buffered LDS filled by GLOBAL_LOAD_ASYNC_TO_LDS_B128.
// ---------------------------------------------------------------------------
#define TK   32
#define LDSP 40   // padded LDS row stride (elements) -> 80B rows, conflict-free frag reads
#define BUFB (128 * LDSP * 2)   // bytes per LDS buffer

__global__ __launch_bounds__(256)
void k_gemm_bf16_wmma(const __bf16* __restrict__ act, long long actStride,
                      const __bf16* __restrict__ wgt, long long wgtStride,
                      float* __restrict__ C, long long cStride,
                      const float* __restrict__ bias, long long biasStride,
                      int M, int N, int K, int beta)
{
  __shared__ __bf16 lsA[2][128 * LDSP];
  __shared__ __bf16 lsW[2][128 * LDSP];

  const int tid  = threadIdx.x;
  const int lane = tid & 31;
  const int wid  = tid >> 5;       // 0..7 (wave32)
  const int wm   = wid & 3;        // M wave group: rows wm*32
  const int wn   = wid >> 2;       // N wave group: cols wn*64
  const int n0   = blockIdx.x * 128;
  const int m0   = blockIdx.y * 128;
  const long long bz = blockIdx.z;

  const __bf16* Ap = act + bz * actStride;
  const __bf16* Wp = wgt + bz * wgtStride;
  float* Cp = C + bz * cStride + (long long)m0 * N + n0;

  // staging map: 256 threads cover 128 rows x 32 k (2 threads/row, 16 bf16 each)
  const int srow  = tid >> 1;
  const int shalf = (tid & 1) * 16;
  const __bf16* gA = Ap + (long long)(m0 + srow) * K + shalf;
  const __bf16* gW = Wp + (long long)(n0 + srow) * K + shalf;
  const unsigned dA = (unsigned)(uintptr_t)&lsA[0][srow * LDSP + shalf];
  const unsigned dW = (unsigned)(uintptr_t)&lsW[0][srow * LDSP + shalf];

  FragC acc[2][4];
  if (beta) {
    #pragma unroll
    for (int mi = 0; mi < 2; ++mi)
      #pragma unroll
      for (int ni = 0; ni < 4; ++ni) {
        const int r0 = wm * 32 + mi * 16 + (lane >> 4) * 8;
        const int c0 = wn * 64 + ni * 16 + (lane & 15);
        #pragma unroll
        for (int e = 0; e < 8; ++e) acc[mi][ni].f[e] = Cp[(long long)(r0 + e) * N + c0];
      }
  } else {
    #pragma unroll
    for (int mi = 0; mi < 2; ++mi)
      #pragma unroll
      for (int ni = 0; ni < 4; ++ni)
        acc[mi][ni].v = (v8f)0.0f;
  }

  // issue one K-tile's staging (4 async b128 per thread: A lo/hi, W lo/hi)
  auto issue_tile = [&](int buf, int kt) {
    const __bf16* ga = gA + (long long)kt * TK;
    const __bf16* gw = gW + (long long)kt * TK;
    async_ld_b128_pair(dA + (unsigned)(buf * BUFB), ga);
    async_ld_b128_pair(dW + (unsigned)(buf * BUFB), gw);
  };

  issue_tile(0, 0);                       // prologue

  const int KT = K / TK;
  for (int kt = 0; kt < KT; ++kt) {
    const bool more = (kt + 1 < KT);
    if (more) issue_tile((kt + 1) & 1, kt + 1);   // prefetch next tile (other buffer)

    // oldest 4 async loads (current tile) complete; next 4 may stay in flight
    if (more) s_wait_async4(); else s_wait_async0();
    __syncthreads();                      // current tile visible to all waves

    const __bf16* la = lsA[kt & 1];
    const __bf16* lw = lsW[kt & 1];

    FragAB aF[2], bF[4];
    const int kb = (lane >> 4) * 16;
    #pragma unroll
    for (int mi = 0; mi < 2; ++mi) {
      const __bf16* p = &la[(wm * 32 + mi * 16 + (lane & 15)) * LDSP + kb];
      aF[mi].q[0] = *(const u32x4*)(p);
      aF[mi].q[1] = *(const u32x4*)(p + 8);
    }
    #pragma unroll
    for (int ni = 0; ni < 4; ++ni) {
      const __bf16* p = &lw[(wn * 64 + ni * 16 + (lane & 15)) * LDSP + kb];
      bF[ni].q[0] = *(const u32x4*)(p);
      bF[ni].q[1] = *(const u32x4*)(p + 8);
    }

    #pragma unroll
    for (int mi = 0; mi < 2; ++mi)
      #pragma unroll
      for (int ni = 0; ni < 4; ++ni)
        acc[mi][ni].v = __builtin_amdgcn_wmma_f32_16x16x32_bf16(
            false, aF[mi].v, false, bF[ni].v, (short)0, acc[mi][ni].v, false, false);

    // all waves done reading this buffer before it is refilled at iter kt+1
    if (more) __syncthreads();
  }

  // epilogue: bias + store f32
  #pragma unroll
  for (int mi = 0; mi < 2; ++mi)
    #pragma unroll
    for (int ni = 0; ni < 4; ++ni) {
      const int r0 = wm * 32 + mi * 16 + (lane >> 4) * 8;
      const int c0 = wn * 64 + ni * 16 + (lane & 15);
      const float bv = bias ? bias[bz * biasStride + n0 + c0] : 0.0f;
      #pragma unroll
      for (int e = 0; e < 8; ++e)
        Cp[(long long)(r0 + e) * N + c0] = acc[mi][ni].f[e] + bv;
    }
}

// ---------------------------------------------------------------------------
// Utility kernels
// ---------------------------------------------------------------------------
__global__ void k_cvt(const float* __restrict__ s, __bf16* __restrict__ d, long long n) {
  long long i = (long long)blockIdx.x * blockDim.x + threadIdx.x;
  const long long stride = (long long)gridDim.x * blockDim.x;
  for (; i < n; i += stride) d[i] = f2bf(s[i]);
}

// Wilc (LA, Z, Z) contracts over its ROW index -> store transposed: d[la][y][z] = s[la][z][y]
__global__ void k_cvt_wilcT(const float* __restrict__ s, __bf16* __restrict__ d,
                            int LA, int Zd) {
  const long long n = (long long)LA * Zd * Zd;
  long long i = (long long)blockIdx.x * blockDim.x + threadIdx.x;
  const long long stride = (long long)gridDim.x * blockDim.x;
  for (; i < n; i += stride) {
    const int y  = (int)(i % Zd);
    const long long r = i / Zd;
    const int z  = (int)(r % Zd);
    const long long la = r / Zd;
    d[(la * Zd + y) * Zd + z] = f2bf(s[i]);
  }
}

__global__ void k_zero(unsigned* __restrict__ p, long long n) {
  long long i = (long long)blockIdx.x * blockDim.x + threadIdx.x;
  const long long stride = (long long)gridDim.x * blockDim.x;
  for (; i < n; i += stride) p[i] = 0u;
}

// LSTM gates for one level: pre (A,B,4,H) f32; updates c state (A,B,H) f32 in place,
// writes new h as bf16 (feeds next level / next timestep / cell_fn GEMMs).
__global__ void k_lstm_gates(const float* __restrict__ pre,
                             float* __restrict__ c_state,
                             __bf16* __restrict__ h_state, int n) {
  const int i = blockIdx.x * blockDim.x + threadIdx.x;
  if (i >= n) return;
  const int hh = i & 511;
  const int ab = i >> 9;
  const float* p = pre + (long long)ab * 2048;
  const float ig = sigf(p[hh]);
  const float fg = sigf(p[512 + hh]);
  const float gg = tanhf(p[1024 + hh]);
  const float og = sigf(p[1536 + hh]);
  const float c  = fg * c_state[i] + ig * gg;
  c_state[i] = c;
  h_state[i] = f2bf(og * tanhf(c));
}

// cell_fn gates: pre (L*A,B,3,Z) f32, sc (L*A*B*Z) f32 -> i_cell/c_cell bf16
__global__ void k_cell_gates(const float* __restrict__ pre,
                             const float* __restrict__ sc,
                             __bf16* __restrict__ icell,
                             __bf16* __restrict__ ccell, int n) {
  const int i = blockIdx.x * blockDim.x + threadIdx.x;
  if (i >= n) return;
  const int z   = i & 511;
  const int lab = i >> 9;
  const float* p = pre + (long long)lab * 1536;
  const float ig = sigf(p[z]);
  const float fg = sigf(p[512 + z]);
  const float gg = tanhf(p[1024 + z]);
  const float ic = ig * sc[i];
  icell[i] = f2bf(ic);
  ccell[i] = f2bf(fg * gg + ic);
}

// combine: per (l,b) row: sum over A, softmax(t1sum) * sigmoid(t2sum), emit cat bf16 (B, L*Z)
__global__ void k_combine(const float* __restrict__ t1, const float* __restrict__ t2,
                          __bf16* __restrict__ cat) {
  const int l = blockIdx.x >> 7;    // L*B blocks, B=128
  const int b = blockIdx.x & 127;
  const int tid = threadIdx.x;      // 256 threads, Z=512 -> 2 per thread
  __shared__ float red[256];

  float s1[2], s2[2];
  #pragma unroll
  for (int j = 0; j < 2; ++j) {
    const int z = tid + j * 256;
    float a1 = 0.f, a2 = 0.f;
    #pragma unroll
    for (int a = 0; a < 3; ++a) {
      const long long base = (((long long)(l * 3 + a) * 128) + b) * 512 + z;
      a1 += t1[base];
      a2 += t2[base];
    }
    s1[j] = a1; s2[j] = a2;
  }
  // row max of s1
  red[tid] = fmaxf(s1[0], s1[1]);
  __syncthreads();
  for (int s = 128; s > 0; s >>= 1) {
    if (tid < s) red[tid] = fmaxf(red[tid], red[tid + s]);
    __syncthreads();
  }
  const float mx = red[0];
  __syncthreads();
  const float e0 = __expf(s1[0] - mx), e1 = __expf(s1[1] - mx);
  red[tid] = e0 + e1;
  __syncthreads();
  for (int s = 128; s > 0; s >>= 1) {
    if (tid < s) red[tid] += red[tid + s];
    __syncthreads();
  }
  const float inv = 1.0f / red[0];
  cat[(long long)b * 2560 + l * 512 + tid]       = f2bf(sigf(s2[0]) * e0 * inv);
  cat[(long long)b * 2560 + l * 512 + tid + 256] = f2bf(sigf(s2[1]) * e1 * inv);
}

// final: y[t][b] = dot(h_new[b], Wlin[t]) + blin[t]; also stash h_new as bf16 carry (h_g)
__global__ void k_final(const float* __restrict__ hnew, const float* __restrict__ wlin,
                        const float* __restrict__ blin, __bf16* __restrict__ hgb,
                        float* __restrict__ y, int t) {
  const int b = blockIdx.x;         // 128 blocks
  const int tid = threadIdx.x;      // 256 threads, H=512
  __shared__ float red[256];
  float acc = 0.f;
  #pragma unroll
  for (int j = 0; j < 2; ++j) {
    const int h = tid + j * 256;
    const float v = hnew[(long long)b * 512 + h];
    hgb[(long long)b * 512 + h] = f2bf(v);
    acc += v * wlin[(long long)t * 512 + h];
  }
  red[tid] = acc;
  __syncthreads();
  for (int s = 128; s > 0; s >>= 1) {
    if (tid < s) red[tid] += red[tid + s];
    __syncthreads();
  }
  if (tid == 0) y[(long long)t * 128 + b] = red[0] + blin[t];
}

// ---------------------------------------------------------------------------
// Host orchestration
// ---------------------------------------------------------------------------
extern "C" void kernel_launch(void* const* d_in, const int* in_sizes, int n_in,
                              void* d_out, int out_size, void* d_ws, size_t ws_size,
                              hipStream_t stream) {
  (void)in_sizes; (void)n_in; (void)out_size; (void)ws_size;
  enum { Tn = 64, Bn = 128, In = 512, Hn = 512, Zn = 512, An = 3, Ln = 5 };

  const float* x    = (const float*)d_in[0];
  const float* Wx   = (const float*)d_in[1];
  const float* Wh   = (const float*)d_in[2];
  const float* bl   = (const float*)d_in[3];
  const float* Wgh  = (const float*)d_in[4];
  const float* Wgp  = (const float*)d_in[5];
  const float* bg   = (const float*)d_in[6];
  const float* Wilc = (const float*)d_in[7];
  const float* bilc = (const float*)d_in[8];
  const float* Wsl  = (const float*)d_in[9];
  const float* bsl  = (const float*)d_in[10];
  const float* Wlin = (const float*)d_in[11];
  const float* blin = (const float*)d_in[12];
  float* y = (float*)d_out;

  // workspace bump allocator (256B aligned regions)
  char* ws = (char*)d_ws;
  size_t off = 0;
  auto alloc = [&](size_t bytes) -> void* {
    void* p = ws + off;
    off += (bytes + 255) & ~(size_t)255;
    return p;
  };

  const long long nWx   = (long long)An * Ln * 4 * Hn * Hn;   // 15,728,640
  const long long nWgh  = (long long)Ln * An * 3 * Zn * Hn;   // 11,796,480
  const long long nWilc = (long long)Ln * An * Zn * Zn;       //  3,932,160
  const long long nWsl  = (long long)Hn * Ln * Zn;            //  1,310,720
  const long long nX    = (long long)Tn * Bn * In;            //  4,194,304
  const long long nState = (long long)Ln * An * Bn * Hn;      //    983,040

  __bf16* wx_bf   = (__bf16*)alloc(nWx * 2);
  __bf16* wh_bf   = (__bf16*)alloc(nWx * 2);
  __bf16* wgh_bf  = (__bf16*)alloc(nWgh * 2);
  __bf16* wgp_bf  = (__bf16*)alloc(nWgh * 2);
  __bf16* wilc_bf = (__bf16*)alloc(nWilc * 2);   // stored transposed
  __bf16* wsl_bf  = (__bf16*)alloc(nWsl * 2);
  __bf16* x_bf    = (__bf16*)alloc(nX * 2);
  __bf16* shb     = (__bf16*)alloc(nState * 2);  // h state (L,A,B,H) bf16
  __bf16* hgb     = (__bf16*)alloc((long long)Bn * Hn * 2);     // h_g carry bf16
  float*  cs      = (float*)alloc(nState * 4);   // c state (L,A,B,H) f32
  float*  preL    = (float*)alloc((long long)An * Bn * 4 * Hn * 4);   // (A,B,2048)
  float*  preC    = (float*)alloc((long long)Ln * An * Bn * 1536 * 4);
  __bf16* icell   = (__bf16*)alloc(nState * 2);
  __bf16* ccell   = (__bf16*)alloc(nState * 2);
  float*  t1      = (float*)alloc(nState * 4);
  float*  t2      = (float*)alloc(nState * 4);
  __bf16* cat     = (__bf16*)alloc((long long)Bn * Ln * Zn * 2);
  float*  hnew    = (float*)alloc((long long)Bn * Hn * 4);

  auto cvt = [&](const float* s, __bf16* d, long long n) {
    int blocks = (int)((n + 255) / 256); if (blocks > 4096) blocks = 4096;
    k_cvt<<<blocks, 256, 0, stream>>>(s, d, n);
  };
  auto zero = [&](void* p, long long bytes) {
    long long n = bytes / 4;
    int blocks = (int)((n + 255) / 256); if (blocks > 2048) blocks = 2048;
    k_zero<<<blocks, 256, 0, stream>>>((unsigned*)p, n);
  };
  auto gemm = [&](const __bf16* act, long long as_, const __bf16* w, long long wsd,
                  float* c, long long cs_, const float* bias, long long bs_,
                  int Mn, int Nn, int Kn, int beta, int batch) {
    dim3 g(Nn / 128, Mn / 128, batch);
    k_gemm_bf16_wmma<<<g, 256, 0, stream>>>(act, as_, w, wsd, c, cs_, bias, bs_,
                                            Mn, Nn, Kn, beta);
  };

  // one-time conversions: bf16 weights total ~118 MB -> resident in 192 MB L2 all run
  cvt(Wx, wx_bf, nWx);
  cvt(Wh, wh_bf, nWx);
  cvt(Wgh, wgh_bf, nWgh);
  cvt(Wgp, wgp_bf, nWgh);
  {
    int blocks = (int)((nWilc + 255) / 256); if (blocks > 4096) blocks = 4096;
    k_cvt_wilcT<<<blocks, 256, 0, stream>>>(Wilc, wilc_bf, Ln * An, Zn);
  }
  cvt(Wsl, wsl_bf, nWsl);
  cvt(x, x_bf, nX);

  // zero-init recurrent state (deterministic per launch)
  zero(shb, nState * 2);
  zero(hgb, (long long)Bn * Hn * 2);
  zero(cs,  nState * 4);

  const long long HH4 = (long long)4 * Hn * Hn;

  for (int t = 0; t < Tn; ++t) {
    // ---- Multi_LSTM: levels sequential, axes batched (grid.z = A) ----
    for (int l = 0; l < Ln; ++l) {
      const __bf16* actx = (l == 0) ? (x_bf + (long long)t * Bn * In)
                                    : (shb + (long long)(l - 1) * An * Bn * Hn);
      const long long actxStride = (l == 0) ? 0 : (long long)Bn * Hn;
      // pre = inp @ Wx^T + b
      gemm(actx, actxStride, wx_bf + (long long)l * HH4, (long long)Ln * HH4,
           preL, (long long)Bn * 2048, bl + (long long)l * 4 * Hn, (long long)Ln * 4 * Hn,
           Bn, 2048, Hn, 0, An);
      // pre += h_prev @ Wh^T
      gemm(shb + (long long)l * An * Bn * Hn, (long long)Bn * Hn,
           wh_bf + (long long)l * HH4, (long long)Ln * HH4,
           preL, (long long)Bn * 2048, nullptr, 0, Bn, 2048, Hn, 1, An);
      // gates -> c (f32, in place), h (bf16)
      k_lstm_gates<<<(An * Bn * Hn) / 256, 256, 0, stream>>>(
          preL, cs + (long long)l * An * Bn * Hn, shb + (long long)l * An * Bn * Hn,
          An * Bn * Hn);
    }

    // ---- cell_fn: batched over L*A = 15 ----
    // pre = h_g @ Wg_h^T + bg   (h_g broadcast: act stride 0)
    gemm(hgb, 0, wgh_bf, (long long)3 * Zn * Hn, preC, (long long)Bn * 1536,
         bg, (long long)3 * Zn, Bn, 1536, Hn, 0, Ln * An);
    // pre += sh @ Wg_p^T
    gemm(shb, (long long)Bn * Hn, wgp_bf, (long long)3 * Zn * Zn, preC,
         (long long)Bn * 1536, nullptr, 0, Bn, 1536, Zn, 1, Ln * An);
    // gates -> i_cell, c_cell bf16
    k_cell_gates<<<(int)(nState / 256), 256, 0, stream>>>(preC, cs, icell, ccell,
                                                          (int)nState);
    // t1 = i_cell @ Wilc + bilc ; t2 = c_cell @ Wilc + bilc  (Wilc pre-transposed)
    gemm(icell, (long long)Bn * Zn, wilc_bf, (long long)Zn * Zn, t1,
         (long long)Bn * Zn, bilc, Zn, Bn, Zn, Zn, 0, Ln * An);
    gemm(ccell, (long long)Bn * Zn, wilc_bf, (long long)Zn * Zn, t2,
         (long long)Bn * Zn, bilc, Zn, Bn, Zn, Zn, 0, Ln * An);
    // sum over A, sigmoid*softmax, build cat (B, L*Z) bf16
    k_combine<<<Ln * Bn, 256, 0, stream>>>(t1, t2, cat);
    // h_new = cat @ Wsl^T + bsl
    gemm(cat, 0, wsl_bf, 0, hnew, 0, bsl, 0, Bn, Hn, Ln * Zn, 0, 1);
    // y[t] = h_new @ Wlin[t]^T + blin[t]; update h_g carry (bf16)
    k_final<<<Bn, 256, 0, stream>>>(hnew, Wlin, blin, hgb, y, t);
  }
}
